// YOLOv5HeadTea_69715909149203
// MI455X (gfx1250) — matrix operand order repaired and checked by reading, not compiled, not run on hardware
//
#include <hip/hip_runtime.h>
#include <hip/hip_bf16.h>

// ---------------------------------------------------------------------------
// Types for CDNA5 WMMA (wave32): v_wmma_f32_16x16x32_bf16
// ---------------------------------------------------------------------------
typedef __attribute__((ext_vector_type(16))) __bf16 v16bf;
typedef __attribute__((ext_vector_type(8)))  __bf16 v8bf;
typedef __attribute__((ext_vector_type(8)))  float  v8f;

#define NBOX_TOTAL 100800
#define NSEG 64

// ---------------------------------------------------------------------------
// Async global->LDS staging (CDNA5 GLOBAL_LOAD_ASYNC_TO_LDS_B128, ASYNCcnt).
// Builtin signature (from compiler diagnostic):
//   (AS1 v4i* global_src, AS3 v4i* lds_dst, imm offset, imm cpol)
// where v4i = int __attribute__((vector_size(16))).
// ---------------------------------------------------------------------------
#if defined(__has_builtin)
#if __has_builtin(__builtin_amdgcn_global_load_async_to_lds_b128)
#define HAS_ASYNC_LDS 1
#endif
#endif
#ifndef HAS_ASYNC_LDS
#define HAS_ASYNC_LDS 0
#endif

#if HAS_ASYNC_LDS
typedef int v4i __attribute__((vector_size(16)));
typedef __attribute__((address_space(1))) v4i* gv4i_ptr;
typedef __attribute__((address_space(3))) v4i* lv4i_ptr;
__device__ __forceinline__ void async_cp16(__bf16* lds_dst, const __bf16* gsrc) {
  __builtin_amdgcn_global_load_async_to_lds_b128(
      (gv4i_ptr)(gsrc), (lv4i_ptr)(lds_dst), 0, 0);
}
__device__ __forceinline__ void wait_async0() {
  asm volatile("s_wait_asynccnt 0x0" ::: "memory");
}
#endif

// ---------------------------------------------------------------------------
// Prep: NCHW fp32 -> position-major bf16  out[pos*C + c]
// ---------------------------------------------------------------------------
__global__ void nchw_to_pbf16(const float* __restrict__ in, __bf16* __restrict__ out,
                              int C, int HW) {
  int g = blockIdx.x * 256 + threadIdx.x;
  if (g >= C * HW) return;
  int c = g / HW;
  int p = g - c * HW;
  out[(size_t)p * C + c] = (__bf16)in[g];
}

__global__ void zero_bf16(__bf16* __restrict__ p, int n) {
  int g = blockIdx.x * 256 + threadIdx.x;
  if (g < n) p[g] = (__bf16)0.0f;
}

// A-fragment K index (contiguous order, matches B staging):
// lanes 0-15:  K=j      (K in {0..15})
// lanes 16-31: K=j+16   (K in {16..31})
__device__ __forceinline__ int a_frag_k(int lane, int j) {
  return (lane < 16) ? j : j + 16;
}

// Pack 3x3 weights (OIHW fp32) into fragment-ordered bf16:
// ap[mtile][kidx][512], kidx = tap*(Cin/32)+kc, element = lane*16 + j
__global__ void pack_w3(const float* __restrict__ w, __bf16* __restrict__ ap,
                        int Cin, int Cout) {
  int e = blockIdx.x * 256 + threadIdx.x;
  int kcN = Cin >> 5;
  int KT = 9 * kcN;
  int total = (Cout >> 4) * KT * 512;
  if (e >= total) return;
  int frag = e >> 9;
  int le = e & 511;
  int kidx = frag % KT;
  int mtile = frag / KT;
  int lane = le >> 4, j = le & 15;
  int t = kidx / kcN, kc = kidx - t * kcN;
  int kk = a_frag_k(lane, j);
  int co = mtile * 16 + (lane & 15);
  int ci = kc * 32 + kk;
  ap[e] = (__bf16)w[(size_t)co * (Cin * 9) + (size_t)ci * 9 + t];
}

// Pack 1x1 weights (255 x Cin fp32) into fragment order, padded to 256 rows.
__global__ void pack_w1(const float* __restrict__ w, __bf16* __restrict__ ap,
                        int Cin, int CoutReal) {
  int e = blockIdx.x * 256 + threadIdx.x;
  int kcN = Cin >> 5;
  int total = 16 * kcN * 512;  // CoutPad=256 -> 16 mtiles
  if (e >= total) return;
  int frag = e >> 9;
  int le = e & 511;
  int kc = frag % kcN;
  int mtile = frag / kcN;
  int lane = le >> 4, j = le & 15;
  int kk = a_frag_k(lane, j);
  int co = mtile * 16 + (lane & 15);
  int ci = kc * 32 + kk;
  ap[e] = (co < CoutReal) ? (__bf16)w[(size_t)co * Cin + ci] : (__bf16)0.0f;
}

// ---------------------------------------------------------------------------
// Implicit-GEMM conv via WMMA bf16.
//   D[co][pos] = sum_{tap,ci} W[co][ci][tap] * feat[ci][pos+shift(tap)]
// Block: 256 thr = 8 waves as 2(M) x 4(N); block tile 128x128; wave 64x32.
// Bt has HW+1 rows: row HW is all-zero; out-of-bounds taps clamp to it, so
// staging is an unconditional 16B/lane copy (async_to_lds-friendly).
// ---------------------------------------------------------------------------
template<int TAPS>
__global__ void __launch_bounds__(256)
conv_gemm_wmma(const __bf16* __restrict__ Apack,  // [Cout/16][KT][512]
               const __bf16* __restrict__ Bt,     // [H*W + 1][Cin], last row zero
               const float*  __restrict__ bias,   // [biasN]
               __bf16* __restrict__ Ct,           // [H*W(+1)][CoutPad]
               int Cin, int CoutPad, int biasN, int H, int W) {
  const int N = H * W;
  const int tid = threadIdx.x;
  const int lane = tid & 31;
  const int wave = tid >> 5;
  const int wm = wave >> 2;        // 0..1 (M)
  const int wn = wave & 3;         // 0..3 (N)
  const int halfSel = lane >> 4;   // 0/1
  const int kcN = Cin >> 5;
  const int KT = TAPS * kcN;
  const int n0 = blockIdx.x * 128;
  const int mt0 = blockIdx.y * 8;  // first 16-row A tile

  // staging-role coords (loop invariant): this thread gathers B fragment `wave`
  const int colS = n0 + wave * 16 + (lane & 15);
  const int pyS = colS / W;
  const int pxS = colS - pyS * W;
  const size_t kOffHalf = (size_t)halfSel * 16;

  v8f acc[4][2];
#pragma unroll
  for (int i = 0; i < 4; ++i)
#pragma unroll
    for (int j = 0; j < 2; ++j)
#pragma unroll
      for (int r = 0; r < 8; ++r) acc[i][j][r] = 0.0f;

#if HAS_ASYNC_LDS
  // -------- double-buffered async pipeline: 1 barrier per K-step ----------
  __shared__ __attribute__((aligned(32))) __bf16 sA[2][4096];
  __shared__ __attribute__((aligned(32))) __bf16 sB[2][4096];

  auto stage = [&](int buf, int kidx, int t, int kc) {
    const __bf16* asrc =
        Apack + ((size_t)(mt0 + wave) * KT + kidx) * 512 + (size_t)lane * 16;
    __bf16* adst = &sA[buf][wave * 512 + lane * 16];
    async_cp16(adst, asrc);
    async_cp16(adst + 8, asrc + 8);
    const int dy = (TAPS == 1) ? 0 : (t / 3 - 1);
    const int dx = (TAPS == 1) ? 0 : (t % 3 - 1);
    const int sy = pyS + dy, sx = pxS + dx;
    const bool okk = (colS < N) && (sy >= 0) && (sy < H) && (sx >= 0) && (sx < W);
    const int srow = okk ? (sy * W + sx) : N;  // N = zero pad row
    const __bf16* bsrc = Bt + (size_t)srow * Cin + (size_t)kc * 32 + kOffHalf;
    __bf16* bdst = &sB[buf][wave * 512 + lane * 16];
    async_cp16(bdst, bsrc);
    async_cp16(bdst + 8, bsrc + 8);
  };

  int tS = 0, kcS = 0;
  stage(0, 0, tS, kcS);
  if (++kcS == kcN) { kcS = 0; ++tS; }

  for (int k = 0; k < KT; ++k) {
    wait_async0();     // my stage-k copies have landed in LDS
    __syncthreads();   // everyone's stage-k copies landed; buf[(k+1)&1] reads done
    if (k + 1 < KT) {
      stage((k + 1) & 1, k + 1, tS, kcS);  // overlaps with WMMAs below
      if (++kcS == kcN) { kcS = 0; ++tS; }
    }
    const __bf16* cA = &sA[k & 1][0];
    const __bf16* cB = &sB[k & 1][0];
    v16bf af[4], bfm[2];
#pragma unroll
    for (int i = 0; i < 4; ++i)
      af[i] = *(const v16bf*)(cA + (wm * 4 + i) * 512 + lane * 16);
#pragma unroll
    for (int j = 0; j < 2; ++j)
      bfm[j] = *(const v16bf*)(cB + (wn * 2 + j) * 512 + lane * 16);
#pragma unroll
    for (int i = 0; i < 4; ++i)
#pragma unroll
      for (int j = 0; j < 2; ++j)
        acc[i][j] = __builtin_amdgcn_wmma_f32_16x16x32_bf16(
            false, af[i], false, bfm[j], (short)0, acc[i][j], false, false);
  }
#else
  // -------- fallback: VGPR-bounce staging, single buffer ------------------
  __shared__ __attribute__((aligned(32))) __bf16 sA1[4096];
  __shared__ __attribute__((aligned(32))) __bf16 sB1[4096];

  for (int t = 0; t < TAPS; ++t) {
    const int dy = (TAPS == 1) ? 0 : (t / 3 - 1);
    const int dx = (TAPS == 1) ? 0 : (t % 3 - 1);
    const int sy = pyS + dy, sx = pxS + dx;
    const bool okk = (colS < N) && (sy >= 0) && (sy < H) && (sx >= 0) && (sx < W);
    const int srow = okk ? (sy * W + sx) : N;  // N = zero pad row
    const __bf16* bsrc = Bt + (size_t)srow * Cin + kOffHalf;

    for (int kc = 0; kc < kcN; ++kc) {
      const int kidx = t * kcN + kc;
      *(v16bf*)(sA1 + wave * 512 + lane * 16) =
          *(const v16bf*)(Apack + ((size_t)(mt0 + wave) * KT + kidx) * 512 + lane * 16);
      *(v16bf*)(sB1 + wave * 512 + lane * 16) =
          *(const v16bf*)(bsrc + (size_t)kc * 32);
      __syncthreads();

      v16bf af[4], bfm[2];
#pragma unroll
      for (int i = 0; i < 4; ++i)
        af[i] = *(const v16bf*)(sA1 + (wm * 4 + i) * 512 + lane * 16);
#pragma unroll
      for (int j = 0; j < 2; ++j)
        bfm[j] = *(const v16bf*)(sB1 + (wn * 2 + j) * 512 + lane * 16);
#pragma unroll
      for (int i = 0; i < 4; ++i)
#pragma unroll
        for (int j = 0; j < 2; ++j)
          acc[i][j] = __builtin_amdgcn_wmma_f32_16x16x32_bf16(
              false, af[i], false, bfm[j], (short)0, acc[i][j], false, false);
      __syncthreads();
    }
  }
#endif

  // Epilogue: C/D layout -> row co = base + (lane>=16)*8 + r, col = lane&15.
  const int coWave = blockIdx.y * 128 + wm * 64;
#pragma unroll
  for (int i = 0; i < 4; ++i) {
    const int coBase = coWave + i * 16 + halfSel * 8;
    float bs[8];
#pragma unroll
    for (int r = 0; r < 8; ++r) {
      int co = coBase + r;
      bs[r] = (co < biasN) ? bias[co] : 0.0f;
    }
#pragma unroll
    for (int j = 0; j < 2; ++j) {
      const int colo = n0 + (wn * 2 + j) * 16 + (lane & 15);
      if (colo < N) {
        v8bf ov;
#pragma unroll
        for (int r = 0; r < 8; ++r) ov[r] = (__bf16)(acc[i][j][r] + bs[r]);
        *(v8bf*)(Ct + (size_t)colo * CoutPad + coBase) = ov;
      }
    }
  }
}

// ---------------------------------------------------------------------------
// Objectness scores (sigmoid of channel a*85+4), lazy decode.
// Global box index g layout: head0 (3*1600), head1 (3*6400), head2 (3*25600).
// ---------------------------------------------------------------------------
__device__ __forceinline__ void box_decompose(int g, int& head, int& a, int& pos,
                                              int& posbase, int& nx, int& ratio) {
  if (g < 4800)       { head = 0; int rel = g;          a = rel / 1600;  pos = rel - a * 1600;  posbase = 0;    nx = 40;  ratio = 32; }
  else if (g < 24000) { head = 1; int rel = g - 4800;   a = rel / 6400;  pos = rel - a * 6400;  posbase = 1600; nx = 80;  ratio = 16; }
  else                { head = 2; int rel = g - 24000;  a = rel / 25600; pos = rel - a * 25600; posbase = 8000; nx = 160; ratio = 8;  }
}

__global__ void obj_scores(const __bf16* __restrict__ detAll, float* __restrict__ scores) {
  int g = blockIdx.x * 256 + threadIdx.x;
  if (g >= NBOX_TOTAL) return;
  int head, a, pos, posbase, nx, ratio;
  box_decompose(g, head, a, pos, posbase, nx, ratio);
  float v = (float)detAll[(size_t)(posbase + pos) * 256 + a * 85 + 4];
  scores[g] = 1.0f / (1.0f + __expf(-v));
}

// ---------------------------------------------------------------------------
// Deterministic top-256: key = (score_bits<<32) | ~idx (ties -> lower index).
// ---------------------------------------------------------------------------
__device__ __forceinline__ unsigned long long make_key(float s, int g) {
  unsigned int bits = __float_as_uint(s);  // scores in (0,1): bits monotonic
  return ((unsigned long long)bits << 32) | (unsigned int)(~(unsigned int)g);
}

__device__ void bsort256(unsigned long long* d, bool ascending) {
  const int t = threadIdx.x;
  for (int k = 2; k <= 256; k <<= 1) {
    for (int j = k >> 1; j > 0; j >>= 1) {
      int ixj = t ^ j;
      if (ixj > t) {
        bool up = ((t & k) == 0) ? ascending : !ascending;
        unsigned long long a = d[t], b = d[ixj];
        if ((a > b) == up) { d[t] = b; d[ixj] = a; }
      }
      __syncthreads();
    }
  }
}

__device__ void bmerge256desc(unsigned long long* d) {
  const int t = threadIdx.x;
  for (int j = 128; j > 0; j >>= 1) {
    int ixj = t ^ j;
    if (ixj > t) {
      unsigned long long a = d[t], b = d[ixj];
      if (a < b) { d[t] = b; d[ixj] = a; }
    }
    __syncthreads();
  }
}

__global__ void topk_stage1(const float* __restrict__ scores, int n,
                            unsigned long long* __restrict__ seglists) {
  __shared__ unsigned long long R[256];
  __shared__ unsigned long long C[256];
  const int t = threadIdx.x;
  const int seglen = (n + NSEG - 1) / NSEG;
  const int s0 = blockIdx.x * seglen;
  const int s1 = (s0 + seglen < n) ? (s0 + seglen) : n;

  int g = s0 + t;
  R[t] = (g < s1) ? make_key(scores[g], g) : 0ull;
  __syncthreads();
  bsort256(R, false);  // descending

  for (int base = s0 + 256; base < s1; base += 256) {
    g = base + t;
    C[t] = (g < s1) ? make_key(scores[g], g) : 0ull;
    __syncthreads();
    bsort256(C, true);  // ascending
    unsigned long long m = (R[t] > C[t]) ? R[t] : C[t];  // desc+asc -> top half
    __syncthreads();
    R[t] = m;
    __syncthreads();
    bmerge256desc(R);
  }
  seglists[blockIdx.x * 256 + t] = R[t];
}

__global__ void topk_stage2(const unsigned long long* __restrict__ seglists,
                            int nseg, int* __restrict__ idx) {
  __shared__ unsigned long long R[256];
  __shared__ unsigned long long C[256];
  const int t = threadIdx.x;
  R[t] = seglists[t];
  __syncthreads();
  for (int s = 1; s < nseg; ++s) {
    C[t] = seglists[s * 256 + (255 - t)];  // reversed -> ascending
    __syncthreads();
    unsigned long long m = (R[t] > C[t]) ? R[t] : C[t];
    __syncthreads();
    R[t] = m;
    __syncthreads();
    bmerge256desc(R);
  }
  idx[t] = (int)(~(unsigned int)(R[t] & 0xffffffffull));
}

// ---------------------------------------------------------------------------
// Gather + decode the 256 selected boxes (85 values each).
// ---------------------------------------------------------------------------
__device__ __constant__ float ANCH[3][3][2] = {
    {{116.f, 90.f}, {156.f, 198.f}, {373.f, 326.f}},   // head0 (mask 6,7,8)
    {{30.f, 61.f},  {62.f, 45.f},   {59.f, 119.f}},    // head1 (mask 3,4,5)
    {{10.f, 13.f},  {16.f, 30.f},   {33.f, 23.f}},     // head2 (mask 0,1,2)
};

__global__ void gather_decode(const int* __restrict__ idx,
                              const __bf16* __restrict__ detAll,
                              float* __restrict__ out) {
  int b = blockIdx.x;
  int o = threadIdx.x;
  if (o >= 85) return;
  int g = idx[b];
  int head, a, pos, posbase, nx, ratio;
  box_decompose(g, head, a, pos, posbase, nx, ratio);
  float v = (float)detAll[(size_t)(posbase + pos) * 256 + a * 85 + o];
  float s = 1.0f / (1.0f + __expf(-v));
  float r;
  if (o == 0)      r = (s * 2.0f - 0.5f + (float)(pos % nx)) * (float)ratio;
  else if (o == 1) r = (s * 2.0f - 0.5f + (float)(pos / nx)) * (float)ratio;
  else if (o < 4)  { float t2 = s * 2.0f; r = t2 * t2 * ANCH[head][a][o - 2]; }
  else             r = s;
  out[b * 85 + o] = r;
}

// ---------------------------------------------------------------------------
// Launch
// ---------------------------------------------------------------------------
extern "C" void kernel_launch(void* const* d_in, const int* in_sizes, int n_in,
                              void* d_out, int out_size, void* d_ws, size_t ws_size,
                              hipStream_t stream) {
  (void)in_sizes; (void)n_in; (void)out_size; (void)ws_size;
  // head i consumes feats reversed: head0->feat2(1024,40), head1->feat1, head2->feat0
  const float* feat[3] = {(const float*)d_in[2], (const float*)d_in[1], (const float*)d_in[0]};
  const float* w3[3] = {(const float*)d_in[3], (const float*)d_in[7],  (const float*)d_in[11]};
  const float* b3[3] = {(const float*)d_in[4], (const float*)d_in[8],  (const float*)d_in[12]};
  const float* w1[3] = {(const float*)d_in[5], (const float*)d_in[9],  (const float*)d_in[13]};
  const float* b1[3] = {(const float*)d_in[6], (const float*)d_in[10], (const float*)d_in[14]};
  const int Cin[3] = {1024, 512, 256};
  const int Hh[3] = {40, 80, 160};
  const int posbase[3] = {0, 1600, 8000};

  char* ws = (char*)d_ws;
  size_t off = 0;
  auto alloc = [&](size_t bytes) -> void* {
    void* p = ws + off;
    off = (off + bytes + 255) & ~(size_t)255;
    return p;
  };

  __bf16* featT[3]; __bf16* ap3[3]; __bf16* ap1[3]; __bf16* yT[3];
  for (int i = 0; i < 3; ++i) {
    int HW = Hh[i] * Hh[i];
    featT[i] = (__bf16*)alloc((size_t)(HW + 1) * Cin[i] * 2);  // +1 zero pad row
    ap3[i]   = (__bf16*)alloc((size_t)Cin[i] * Cin[i] * 9 * 2);
    ap1[i]   = (__bf16*)alloc((size_t)256 * Cin[i] * 2);
    yT[i]    = (__bf16*)alloc((size_t)(HW + 1) * Cin[i] * 2);  // +1 zero pad row
  }
  __bf16* detAll = (__bf16*)alloc((size_t)33600 * 256 * 2);
  float* scores = (float*)alloc((size_t)NBOX_TOTAL * 4);
  unsigned long long* seglists = (unsigned long long*)alloc((size_t)NSEG * 256 * 8);
  int* idx = (int*)alloc(256 * 4);

  for (int i = 0; i < 3; ++i) {
    const int H = Hh[i], W = Hh[i], HW = H * W, C = Cin[i];
    nchw_to_pbf16<<<dim3((C * HW + 255) / 256), dim3(256), 0, stream>>>(feat[i], featT[i], C, HW);
    zero_bf16<<<dim3((C + 255) / 256), dim3(256), 0, stream>>>(featT[i] + (size_t)HW * C, C);
    zero_bf16<<<dim3((C + 255) / 256), dim3(256), 0, stream>>>(yT[i] + (size_t)HW * C, C);
    pack_w3<<<dim3((C * C * 9 + 255) / 256), dim3(256), 0, stream>>>(w3[i], ap3[i], C, C);
    pack_w1<<<dim3((256 * C + 255) / 256), dim3(256), 0, stream>>>(w1[i], ap1[i], C, 255);

    dim3 g3((HW + 127) / 128, C / 128);
    conv_gemm_wmma<9><<<g3, dim3(256), 0, stream>>>(ap3[i], featT[i], b3[i], yT[i], C, C, C, H, W);

    dim3 g1((HW + 127) / 128, 2);  // CoutPad=256 -> 2 row-blocks of 128
    conv_gemm_wmma<1><<<g1, dim3(256), 0, stream>>>(ap1[i], yT[i], b1[i],
                                                    detAll + (size_t)posbase[i] * 256,
                                                    C, 256, 255, H, W);
  }

  obj_scores<<<dim3((NBOX_TOTAL + 255) / 256), dim3(256), 0, stream>>>(detAll, scores);
  topk_stage1<<<dim3(NSEG), dim3(256), 0, stream>>>(scores, NBOX_TOTAL, seglists);
  topk_stage2<<<dim3(1), dim3(256), 0, stream>>>(seglists, NSEG, idx);
  gather_decode<<<dim3(256), dim3(128), 0, stream>>>(idx, detAll, (float*)d_out);
}
